// RNNTJoint_84447646974635
// MI455X (gfx1250) — matrix operand order repaired and checked by reading, not compiled, not run on hardware
//
#include <hip/hip_runtime.h>

// ---------------------------------------------------------------------------
// RNNT joint network, fused for MI455X (gfx1250, wave32, WMMA).
//   e = enc^T @ W_enc + b_enc           [B,T,JH]
//   p = dec^T @ W_pred + b_pred         [B,U,JH]
//   out = log_softmax(relu(e+p) @ W_out + b_out)   [B,T,U,NC]
// Main GEMM (137 GFLOP) runs on v_wmma_f32_16x16x32_bf16; softmax is fused
// (online/flash style) so the 537 MB output is written exactly once.
// ---------------------------------------------------------------------------

typedef __attribute__((ext_vector_type(16))) __bf16 v16bf;
typedef __attribute__((ext_vector_type(8)))  float  v8f;

#define B_   4
#define T_   256
#define U_   128
#define EH_  512
#define PH_  320
#define JH_  512
#define NC_  1025
#define NTILE 65      // ceil(NC/16)
#define KSTEPS 16     // JH / 32
#define MAXT_PER_WAVE 17  // ceil(NTILE/4)

__device__ __forceinline__ unsigned short f2bf(float f) {
  unsigned u = __float_as_uint(f);
  u += 0x7fffu + ((u >> 16) & 1u);          // round-to-nearest-even
  return (unsigned short)(u >> 16);
}

// out[b][t][j] = sum_h in[b][h][t] * W[h][j] + bias[j]
__global__ void proj_kernel(const float* __restrict__ in, const float* __restrict__ W,
                            const float* __restrict__ bias, float* __restrict__ out,
                            int H, int Tdim) {
  int tid = blockIdx.x * blockDim.x + threadIdx.x;
  int j  = tid & (JH_ - 1);
  int bt = tid >> 9;
  int t  = bt % Tdim;
  int b  = bt / Tdim;
  const float* col = in + ((size_t)b * H) * Tdim + t;     // stride Tdim (broadcast per j-group)
  float acc = bias[j];
#pragma unroll 4
  for (int h = 0; h < H; ++h)
    acc = fmaf(col[(size_t)h * Tdim], W[(size_t)h * JH_ + j], acc);
  out[tid] = acc;
}

// Pre-swizzle W_out -> bf16, WMMA B-fragment-major:
//   Bp[((nt*16 + k)*32 + lane)*16 + e]   (each lane's 16 bf16 contiguous = 32B)
//   lanes 0-15:  N = lane,    K = k*32 + e        (e = 0..15)
//   lanes 16-31: N = lane-16, K = k*32 + 16 + e
__global__ void pack_wout(const float* __restrict__ Wo, unsigned short* __restrict__ Bp) {
  int tid  = blockIdx.x * blockDim.x + threadIdx.x;       // 65*16*32*16 = 532480
  int e    = tid & 15;
  int lane = (tid >> 4) & 31;
  int k    = (tid >> 9) & 15;
  int nt   = tid >> 13;
  int c = nt * 16 + (lane & 15);
  int j = k * 32 + ((lane >> 4) << 4) + e;
  float v = (c < NC_) ? Wo[(size_t)j * NC_ + c] : 0.0f;   // zero-pad NC 1025 -> 1040
  Bp[tid] = f2bf(v);
}

// One block = 16 output rows (fixed (b,t), u0..u0+15), full NC, fused softmax.
__global__ __launch_bounds__(128, 1)
void joint_kernel(const float* __restrict__ e, const float* __restrict__ p,
                  const unsigned short* __restrict__ Bp,
                  const float* __restrict__ b_out, float* __restrict__ out) {
  // A tile (relu(e+p)) as bf16; +8 pad ushorts/row => 1040B row stride, conflict-free b128 reads
  __shared__ __align__(16) unsigned short hA[16][JH_ + 8];
  __shared__ float redM[4][16];
  __shared__ float redS[4][16];
  __shared__ float shiftS[16];

  const int tid = threadIdx.x;
  const int r0  = blockIdx.x * 16;          // global row tile start
  const int u0  = r0 & (U_ - 1);
  const int bt  = r0 / U_;                  // = b*T + t
  const int b   = bt >> 8;                  // T_ = 256
  const float* erow = e + (size_t)bt * JH_;
  const float* prow = p + ((size_t)b * U_ + u0) * JH_;

  // ---- build 16 x 512 bf16 A tile in LDS ----
  for (int idx = tid; idx < 16 * (JH_ / 4); idx += 128) {
    int row = idx >> 7;                     // JH_/4 = 128
    int j4  = (idx & 127) << 2;
    const float4 ev = *(const float4*)(erow + j4);
    const float4 pv = *(const float4*)(prow + (size_t)row * JH_ + j4);
    unsigned short* dst = &hA[row][j4];
    dst[0] = f2bf(fmaxf(ev.x + pv.x, 0.f));
    dst[1] = f2bf(fmaxf(ev.y + pv.y, 0.f));
    dst[2] = f2bf(fmaxf(ev.z + pv.z, 0.f));
    dst[3] = f2bf(fmaxf(ev.w + pv.w, 0.f));
  }
  __syncthreads();

  const int wave   = tid >> 5;
  const int lane   = tid & 31;
  const int nlo    = lane & 15;             // N within tile / A row M
  const int hiHalf = lane >> 4;
  const int koffA  = hiHalf << 4;           // A frag K sub-offset: 0 or 16
  const int mbase  = hiHalf << 3;           // C rows: 0..7 or 8..15

  v8f   acc[MAXT_PER_WAVE];
  float bo[MAXT_PER_WAVE];
  float m[8], s[8];
#pragma unroll
  for (int v = 0; v < 8; ++v) { m[v] = -3.0e38f; s[v] = 0.f; }

  // ---- GEMM over this wave's n-tiles, accumulators stay in registers ----
#pragma unroll
  for (int i = 0; i < MAXT_PER_WAVE; ++i) {
    const int nt = wave + (i << 2);
    v8f z = {};
    acc[i] = z;
    bo[i]  = 0.f;
    if (nt < NTILE) {
      const int  c      = nt * 16 + nlo;
      const bool cvalid = (c < NC_);
      bo[i] = cvalid ? b_out[c] : 0.f;
      const unsigned short* bp = Bp + ((size_t)nt << 13) + ((size_t)lane << 4);
      v8f cacc = {};
      for (int k = 0; k < KSTEPS; ++k) {
        union { v16bf v; uint4 q[2]; } A, Bf;
        const unsigned short* ap = &hA[nlo][k * 32 + koffA];
        A.q[0] = *(const uint4*)ap;              // K = k*32 + koffA + 0..7
        A.q[1] = *(const uint4*)(ap + 8);        // K = k*32 + koffA + 8..15
        const unsigned short* bk = bp + ((size_t)k << 9);
        Bf.q[0] = *(const uint4*)bk;
        Bf.q[1] = *(const uint4*)(bk + 8);
        __builtin_prefetch(bk + 512, 0, 1);      // next k-step's B frag (L2 resident)
        cacc = __builtin_amdgcn_wmma_f32_16x16x32_bf16(
                   false, A.v, false, Bf.v, (short)0, cacc, false, false);
      }
      acc[i] = cacc;
      if (cvalid) {                              // online softmax stats per row
#pragma unroll
        for (int v = 0; v < 8; ++v) {
          float x  = cacc[v] + bo[i];
          float mn = fmaxf(m[v], x);
          s[v] = s[v] * __expf(m[v] - mn) + __expf(x - mn);
          m[v] = mn;
        }
      }
    }
  }

  // ---- reduce (m,s) across the 16 lanes of each half-wave ----
#pragma unroll
  for (int off = 1; off < 16; off <<= 1) {
#pragma unroll
    for (int v = 0; v < 8; ++v) {
      float mo = __shfl_xor(m[v], off, 32);
      float so = __shfl_xor(s[v], off, 32);
      float mn = fmaxf(m[v], mo);
      s[v] = s[v] * __expf(m[v] - mn) + so * __expf(mo - mn);
      m[v] = mn;
    }
  }
  if (nlo == 0) {
#pragma unroll
    for (int v = 0; v < 8; ++v) { redM[wave][mbase + v] = m[v]; redS[wave][mbase + v] = s[v]; }
  }
  __syncthreads();
  // ---- merge across 4 waves, row shift = max + log(sum) ----
  if (tid < 16) {
    float mm = redM[0][tid], ss = redS[0][tid];
#pragma unroll
    for (int w = 1; w < 4; ++w) {
      float mo = redM[w][tid], so = redS[w][tid];
      float mn = fmaxf(mm, mo);
      ss = ss * __expf(mm - mn) + so * __expf(mo - mn);
      mm = mn;
    }
    shiftS[tid] = mm + __logf(ss);
  }
  __syncthreads();

  float sh[8];
#pragma unroll
  for (int v = 0; v < 8; ++v) sh[v] = shiftS[mbase + v];

  // ---- replay registers -> final log-softmax, single output write ----
#pragma unroll
  for (int i = 0; i < MAXT_PER_WAVE; ++i) {
    const int nt = wave + (i << 2);
    if (nt < NTILE) {
      const int c = nt * 16 + nlo;
      if (c < NC_) {
        float* orow = out + (size_t)(r0 + mbase) * NC_ + c;
#pragma unroll
        for (int v = 0; v < 8; ++v)
          orow[(size_t)v * NC_] = acc[i][v] + bo[i] - sh[v];
      }
    }
  }
}

extern "C" void kernel_launch(void* const* d_in, const int* in_sizes, int n_in,
                              void* d_out, int out_size, void* d_ws, size_t ws_size,
                              hipStream_t stream) {
  const float* enc    = (const float*)d_in[0];
  const float* dec    = (const float*)d_in[1];
  const float* W_enc  = (const float*)d_in[2];
  const float* b_enc  = (const float*)d_in[3];
  const float* W_pred = (const float*)d_in[4];
  const float* b_pred = (const float*)d_in[5];
  const float* W_out  = (const float*)d_in[6];
  const float* b_out  = (const float*)d_in[7];
  float* out = (float*)d_out;

  // workspace layout (bytes): e f32 [0, 2MiB) | p f32 [2MiB, 3MiB) | Bp bf16 (~1.02MiB)
  char* ws = (char*)d_ws;
  float*          e_ws = (float*)(ws);
  float*          p_ws = (float*)(ws + 2097152);
  unsigned short* Bp   = (unsigned short*)(ws + 2097152 + 1048576);

  proj_kernel<<<(B_ * T_ * JH_) / 256, 256, 0, stream>>>(enc, W_enc, b_enc, e_ws, EH_, T_);
  proj_kernel<<<(B_ * U_ * JH_) / 256, 256, 0, stream>>>(dec, W_pred, b_pred, p_ws, PH_, U_);
  pack_wout<<<(NTILE * 16 * 32 * 16) / 256, 256, 0, stream>>>(W_out, Bp);
  joint_kernel<<<(B_ * T_ * U_) / 16, 128, 0, stream>>>(e_ws, p_ws, Bp, b_out, out);
}
